// VRNN_75368086110979
// MI455X (gfx1250) — matrix operand order repaired and checked by reading, not compiled, not run on hardware
//
#include <hip/hip_runtime.h>
#include <hip/hip_bf16.h>
#include <cmath>

// ---------------------------------------------------------------------------
// CDNA5 (gfx1250) VRNN forward: f16 WMMA GEMMs with f32 accumulate.
// All matrix ops via v_wmma_f32_16x16x32_f16 (wave32); operands fed with
// 128-bit LDS/global loads by laying data out in a contiguous K=hs*16+j fragment order.
// ---------------------------------------------------------------------------

typedef __attribute__((ext_vector_type(16))) _Float16 v16h;
typedef __attribute__((ext_vector_type(8)))  _Float16 v8h;
typedef __attribute__((ext_vector_type(8)))  float    v8f;

#define ACT_NONE 0
#define ACT_RELU 1
#define ACT_SP   2

__device__ __forceinline__ v8f wmma16(v16h a, v16h b, v8f c) {
  return __builtin_amdgcn_wmma_f32_16x16x32_f16(false, a, false, b, (short)0, c,
                                                false, false);
}
__device__ __forceinline__ v16h cat8(v8h lo, v8h hi) {
  return __builtin_shufflevector(lo, hi, 0, 1, 2, 3, 4, 5, 6, 7, 8, 9, 10, 11,
                                 12, 13, 14, 15);
}
__device__ __forceinline__ float sp_act(float x) {
  return fmaxf(x, 0.f) + log1pf(expf(-fabsf(x)));  // stable softplus
}
__device__ __forceinline__ float sigm(float x) { return 1.f / (1.f + expf(-x)); }
__device__ __forceinline__ float apply_act(float v, int act) {
  if (act == ACT_RELU) return fmaxf(v, 0.f);
  if (act == ACT_SP)   return sp_act(v);
  return v;
}

// ---------------------------------------------------------------------------
// Fragment layouts (as used in this file, wave32, 16-bit):
//  A 16x32: lane = hs*16+m; elem j -> K=hs*16+j
//  B 32x16: lane = hs*16+n; elem j -> K=hs*16+j
//  C 16x16 f32: vgpr r, lane hs*16+n -> (m=hs*8+r, n)
// ---------------------------------------------------------------------------

// A fragment from f16 row-major (global or LDS; lda in halfs, multiple of 8).
__device__ __forceinline__ v16h a_frag16(const _Float16* A, int lda, int m0,
                                         int k0, int lane) {
  const _Float16* row =
      A + (size_t)(m0 + (lane & 15)) * lda + k0 + ((lane >> 4) << 4);
  return cat8(*(const v8h*)row, *(const v8h*)(row + 8));
}
// B fragment from pre-swizzled block layout: block(kt,nt) = 512 halfs,
// lane L reads halfs [L*16, L*16+16).
__device__ __forceinline__ v16h b_frag_p(const _Float16* Bp, int NT, int kt,
                                         int nt, int lane) {
  return *(const v16h*)(Bp + (((size_t)kt * NT + nt) << 9) + (lane << 4));
}
__device__ __forceinline__ void zero4(v8f a[4]) {
#pragma unroll
  for (int c = 0; c < 4; ++c)
#pragma unroll
    for (int r = 0; r < 8; ++r) a[c][r] = 0.f;
}

// acc[c] += A[m0..m0+15, 0..K) x Bp(:, tiles nt0+c), 4 col tiles
__device__ __forceinline__ void mm4(v8f acc[4], const _Float16* A, int lda,
                                    int m0, const _Float16* Bp, int NT, int nt0,
                                    int ktb, int K, int lane) {
  for (int k = 0; k < K; k += 32) {
    v16h a = a_frag16(A, lda, m0, k, lane);
    int kt = ktb + (k >> 5);
#pragma unroll
    for (int c = 0; c < 4; ++c)
      acc[c] = wmma16(a, b_frag_p(Bp, NT, kt, nt0 + c, lane), acc[c]);
  }
}
// dual-B with shared A (enc_mean / enc_std)
__device__ __forceinline__ void mm8(v8f am[4], v8f as[4], const _Float16* A,
                                    int lda, int m0, const _Float16* B0,
                                    const _Float16* B1, int NT, int nt0, int K,
                                    int lane) {
  for (int k = 0; k < K; k += 32) {
    v16h a = a_frag16(A, lda, m0, k, lane);
    int kt = k >> 5;
#pragma unroll
    for (int c = 0; c < 4; ++c) {
      am[c] = wmma16(a, b_frag_p(B0, NT, kt, nt0 + c, lane), am[c]);
      as[c] = wmma16(a, b_frag_p(B1, NT, kt, nt0 + c, lane), as[c]);
    }
  }
}
// 3 gate groups (r,z,n at nt offsets 0/8/16) with shared A; NT must be 24.
__device__ __forceinline__ void mm12(v8f aR[4], v8f aZ[4], v8f aN[4],
                                     const _Float16* A, int lda, int m0,
                                     const _Float16* Bp, int nt0, int ktb,
                                     int K, int lane) {
  for (int k = 0; k < K; k += 32) {
    v16h a = a_frag16(A, lda, m0, k, lane);
    int kt = ktb + (k >> 5);
#pragma unroll
    for (int c = 0; c < 4; ++c)
      aR[c] = wmma16(a, b_frag_p(Bp, 24, kt, nt0 + c, lane), aR[c]);
#pragma unroll
    for (int c = 0; c < 4; ++c)
      aZ[c] = wmma16(a, b_frag_p(Bp, 24, kt, 8 + nt0 + c, lane), aZ[c]);
#pragma unroll
    for (int c = 0; c < 4; ++c)
      aN[c] = wmma16(a, b_frag_p(Bp, 24, kt, 16 + nt0 + c, lane), aN[c]);
  }
}

// ---------------------------------------------------------------------------
// Weight pre-swizzle: f32 [K,N] row-major -> f16 fragment-order blocks.
// Requires K%32==0, N%16==0 (true for all recurrent weights).
// ---------------------------------------------------------------------------
__global__ void pack_b_kernel(const float* __restrict__ B,
                              _Float16* __restrict__ Bp, int K, int N) {
  int i = blockIdx.x * blockDim.x + threadIdx.x;
  if (i >= K * N) return;
  int blk = i >> 9, r = i & 511;
  int lane = r >> 4, j = r & 15;
  int NT = N >> 4;
  int kt = blk / NT, nt = blk - kt * NT;
  int k = kt * 32 + ((lane >> 4) << 4) + j;
  int n = nt * 16 + (lane & 15);
  Bp[i] = (_Float16)B[(size_t)k * N + n];
}

// ---------------------------------------------------------------------------
// Shared tile-GEMM building blocks (128x128 tile, K-step 32, 8 waves).
// As: row-major (k contiguous), Bt: transposed (Bt[n][k]); stride 40 halfs
// (80B rows: 16B-aligned, bank-spread).
// ---------------------------------------------------------------------------
__device__ __forceinline__ void stage_a_fast(const float* ap, int arow, int ac0,
                                             _Float16 (*As)[40]) {
  v8h lo, hi;
#pragma unroll
  for (int j = 0; j < 8; ++j) {
    lo[j] = (_Float16)ap[j];
    hi[j] = (_Float16)ap[8 + j];
  }
  *(v8h*)&As[arow][ac0] = lo;
  *(v8h*)&As[arow][ac0 + 8] = hi;
}
__device__ __forceinline__ void stage_a_edge(const float* ap, int gm, int M,
                                             int kg, int K, int arow, int ac0,
                                             _Float16 (*As)[40]) {
#pragma unroll
  for (int j = 0; j < 16; ++j)
    As[arow][ac0 + j] =
        (gm < M && kg + j < K) ? (_Float16)ap[j] : (_Float16)0.f;
}
__device__ __forceinline__ void stage_bt(const float* bp, bool krow_ok,
                                         bool nfull, int n0, int N, int brow,
                                         int bc0, _Float16 (*Bt)[40]) {
  if (krow_ok) {
    if (nfull) {
#pragma unroll
      for (int j = 0; j < 16; ++j) Bt[bc0 + j][brow] = (_Float16)bp[j];
    } else {
#pragma unroll
      for (int j = 0; j < 16; ++j)
        Bt[bc0 + j][brow] =
            (n0 + bc0 + j < N) ? (_Float16)bp[j] : (_Float16)0.f;
    }
  } else {
#pragma unroll
    for (int j = 0; j < 16; ++j) Bt[bc0 + j][brow] = (_Float16)0.f;
  }
}
__device__ __forceinline__ void load_af(const _Float16 (*As)[40], int wm,
                                        int lane, v16h af[2]) {
  const int mm = lane & 15, hs8 = (lane >> 4) << 4;
#pragma unroll
  for (int i = 0; i < 2; ++i) {
    const _Float16* p = &As[wm * 32 + i * 16 + mm][hs8];
    af[i] = cat8(*(const v8h*)p, *(const v8h*)(p + 8));
  }
}
__device__ __forceinline__ void tile_mma(const _Float16 (*Bt)[40],
                                         const v16h af[2], v8f acc[2][4],
                                         int wn, int lane) {
  const int nn = lane & 15, kh = (lane >> 4) << 4;
#pragma unroll
  for (int j4 = 0; j4 < 4; ++j4) {
    const _Float16* q = &Bt[wn * 64 + j4 * 16 + nn][kh];
    v16h bf = cat8(*(const v8h*)q, *(const v8h*)(q + 8));
    acc[0][j4] = wmma16(af[0], bf, acc[0][j4]);
    acc[1][j4] = wmma16(af[1], bf, acc[1][j4]);
  }
}
__device__ __forceinline__ void epilogue(const v8f acc[2][4],
                                         const float* __restrict__ bias,
                                         float* __restrict__ C,
                                         _Float16* __restrict__ C16, int M,
                                         int N, int ldc, int act, int m0base,
                                         int n0base, int wm, int wn, int lane) {
  const int nn = lane & 15, hs = lane >> 4;
#pragma unroll
  for (int i = 0; i < 2; ++i) {
#pragma unroll
    for (int j = 0; j < 4; ++j) {
      int gn = n0base + wn * 64 + j * 16 + nn;
      if (gn >= N) continue;
      float bv = bias[gn];
      int gmb = m0base + wm * 32 + i * 16 + hs * 8;
#pragma unroll
      for (int r = 0; r < 8; ++r) {
        int gm = gmb + r;
        if (gm >= M) continue;
        float v = apply_act(acc[i][j][r] + bv, act);
        if (C)   C[(size_t)gm * ldc + gn] = v;
        if (C16) C16[(size_t)gm * ldc + gn] = (_Float16)v;
      }
    }
  }
}

// ---------------------------------------------------------------------------
// Single-B GEMM: act(A[M,K] @ B[K,N] + bias) -> C (f32) and/or C16 (f16).
// K-tail peeled out of the main loop (branch-free staging in steady state).
// ---------------------------------------------------------------------------
__global__ __launch_bounds__(256) void gemm_bias_act_kernel(
    const float* __restrict__ A, const float* __restrict__ B,
    const float* __restrict__ bias, float* __restrict__ C,
    _Float16* __restrict__ C16, int M, int N, int K, int lda, int ldb, int ldc,
    int act) {
  __shared__ _Float16 As[128][40];
  __shared__ _Float16 Bt[128][40];

  const int tid = threadIdx.x;
  const int lane = tid & 31, w = tid >> 5;
  const int wm = w & 3, wn = w >> 2;
  const int m0base = blockIdx.y * 128;
  const int n0base = blockIdx.x * 128;

  v8f acc[2][4];
#pragma unroll
  for (int i = 0; i < 2; ++i) zero4(acc[i]);

  const int arow = tid >> 1, ac0 = (tid & 1) * 16;  // A tile 128x32
  const int brow = tid >> 3, bc0 = (tid & 7) * 16;  // B tile 32x128
  const bool mful = (m0base + 128 <= M);
  const bool nful = (n0base + 128 <= N);
  const int Ktail = K & 31;
  const int Kmain = K - Ktail;

  for (int k0 = 0; k0 < Kmain; k0 += 32) {
    const float* ap = A + (size_t)(m0base + arow) * lda + k0 + ac0;
    if (mful) stage_a_fast(ap, arow, ac0, As);
    else      stage_a_edge(ap, m0base + arow, M, k0 + ac0, K, arow, ac0, As);
    stage_bt(B + (size_t)(k0 + brow) * ldb + n0base + bc0, true, nful, n0base,
             N, brow, bc0, Bt);
    __syncthreads();
    v16h af[2];
    load_af(As, wm, lane, af);
    tile_mma(Bt, af, acc, wn, lane);
    __syncthreads();
  }
  if (Ktail) {
    const int k0 = Kmain;
    const float* ap = A + (size_t)(m0base + arow) * lda + k0 + ac0;
    stage_a_edge(ap, m0base + arow, M, k0 + ac0, K, arow, ac0, As);
    stage_bt(B + (size_t)(k0 + brow) * ldb + n0base + bc0, k0 + brow < K, nful,
             n0base, N, brow, bc0, Bt);
    __syncthreads();
    v16h af[2];
    load_af(As, wm, lane, af);
    tile_mma(Bt, af, acc, wn, lane);
    __syncthreads();
  }
  epilogue(acc, bias, C, C16, M, N, ldc, act, m0base, n0base, wm, wn, lane);
}

// ---------------------------------------------------------------------------
// Dual-B GEMM (fused mean/std heads): A staged once, two B tiles, 16 WMMAs
// per K-step, two epilogues. Halves A traffic for the big decoder heads.
// ---------------------------------------------------------------------------
__global__ __launch_bounds__(256) void gemm_dual_kernel(
    const float* __restrict__ A, const float* __restrict__ B0,
    const float* __restrict__ B1, const float* __restrict__ bias0,
    const float* __restrict__ bias1, float* __restrict__ C0,
    float* __restrict__ C1, int M, int N, int K, int lda, int ldb, int ldc,
    int act0, int act1) {
  __shared__ _Float16 As[128][40];
  __shared__ _Float16 Bt0[128][40];
  __shared__ _Float16 Bt1[128][40];

  const int tid = threadIdx.x;
  const int lane = tid & 31, w = tid >> 5;
  const int wm = w & 3, wn = w >> 2;
  const int m0base = blockIdx.y * 128;
  const int n0base = blockIdx.x * 128;

  v8f acc0[2][4], acc1[2][4];
#pragma unroll
  for (int i = 0; i < 2; ++i) {
    zero4(acc0[i]);
    zero4(acc1[i]);
  }

  const int arow = tid >> 1, ac0 = (tid & 1) * 16;
  const int brow = tid >> 3, bc0 = (tid & 7) * 16;
  const bool mful = (m0base + 128 <= M);
  const bool nful = (n0base + 128 <= N);
  const int Ktail = K & 31;
  const int Kmain = K - Ktail;

  for (int k0 = 0; k0 < Kmain; k0 += 32) {
    const float* ap = A + (size_t)(m0base + arow) * lda + k0 + ac0;
    if (mful) stage_a_fast(ap, arow, ac0, As);
    else      stage_a_edge(ap, m0base + arow, M, k0 + ac0, K, arow, ac0, As);
    stage_bt(B0 + (size_t)(k0 + brow) * ldb + n0base + bc0, true, nful, n0base,
             N, brow, bc0, Bt0);
    stage_bt(B1 + (size_t)(k0 + brow) * ldb + n0base + bc0, true, nful, n0base,
             N, brow, bc0, Bt1);
    __syncthreads();
    v16h af[2];
    load_af(As, wm, lane, af);
    tile_mma(Bt0, af, acc0, wn, lane);
    tile_mma(Bt1, af, acc1, wn, lane);
    __syncthreads();
  }
  if (Ktail) {
    const int k0 = Kmain;
    const float* ap = A + (size_t)(m0base + arow) * lda + k0 + ac0;
    stage_a_edge(ap, m0base + arow, M, k0 + ac0, K, arow, ac0, As);
    stage_bt(B0 + (size_t)(k0 + brow) * ldb + n0base + bc0, k0 + brow < K,
             nful, n0base, N, brow, bc0, Bt0);
    stage_bt(B1 + (size_t)(k0 + brow) * ldb + n0base + bc0, k0 + brow < K,
             nful, n0base, N, brow, bc0, Bt1);
    __syncthreads();
    v16h af[2];
    load_af(As, wm, lane, af);
    tile_mma(Bt0, af, acc0, wn, lane);
    tile_mma(Bt1, af, acc1, wn, lane);
    __syncthreads();
  }
  epilogue(acc0, bias0, C0, nullptr, M, N, ldc, act0, m0base, n0base, wm, wn, lane);
  epilogue(acc1, bias1, C1, nullptr, M, N, ldc, act1, m0base, n0base, wm, wn, lane);
}

// ---------------------------------------------------------------------------
// Sequential VRNN scan: one workgroup (8 waves), 64 steps, B=64, H=Z=128.
// h/t1/t2/pz in LDS f16; weights pre-swizzled f16 in global (L2-resident);
// phi_all pre-converted f16. All GEMMs via WMMA with b128 operand loads.
// Wave w: rows [16*(w>>1),+16), col tiles nt0 = 4*(w&1) .. +4.
// ---------------------------------------------------------------------------
__global__ __launch_bounds__(256) void vrnn_scan_kernel(
    const _Float16* __restrict__ phi16,  // [4096, 384] f16
    const float* __restrict__ eps,       // [4096, 128]
    const _Float16* __restrict__ enc1p, const float* __restrict__ enc1b,
    const _Float16* __restrict__ enc2p, const float* __restrict__ enc2b,
    const _Float16* __restrict__ emp,   const float* __restrict__ emb,
    const _Float16* __restrict__ esp,   const float* __restrict__ esb,
    const _Float16* __restrict__ pzp,   const float* __restrict__ pzb,
    const _Float16* __restrict__ gihp,  const _Float16* __restrict__ ghhp,
    float* __restrict__ out_im, float* __restrict__ out_istd,
    float* __restrict__ h_prev_all, float* __restrict__ dec_in_all) {
  __shared__ _Float16 h16[64][128];
  __shared__ _Float16 t1[64][128];
  __shared__ _Float16 t2[64][128];
  __shared__ _Float16 pz16[64][128];

  const int tid = threadIdx.x;
  const int lane = tid & 31, w = tid >> 5;
  const int m0 = (w >> 1) * 16;
  const int nt0 = (w & 1) * 4;
  const int n0 = nt0 * 16;
  const int nn = lane & 15, hs = lane >> 4;

  for (int i = tid; i < 64 * 128; i += 256) ((_Float16*)h16)[i] = (_Float16)0.f;
  __syncthreads();

  for (int t = 0; t < 64; ++t) {
    const _Float16* phiA = phi16 + (size_t)t * 64 * 384;
    const float* epsT = eps + (size_t)t * 64 * 128;
    const size_t obase = (size_t)t * 64 * 128;

    // ---- enc1: relu([phi | h] @ W + b) -> t1 -----------------------------
    v8f acc[4];
    zero4(acc);
    mm4(acc, phiA, 384, m0, enc1p, 8, nt0, 0, 384, lane);
    mm4(acc, &h16[0][0], 128, m0, enc1p, 8, nt0, 12, 128, lane);
#pragma unroll
    for (int c = 0; c < 4; ++c) {
      int n = n0 + c * 16 + nn;
      float bv = enc1b[n];
#pragma unroll
      for (int r = 0; r < 8; ++r)
        t1[m0 + hs * 8 + r][n] = (_Float16)fmaxf(acc[c][r] + bv, 0.f);
    }
    __syncthreads();

    // ---- enc2 -> t2 ------------------------------------------------------
    zero4(acc);
    mm4(acc, &t1[0][0], 128, m0, enc2p, 8, nt0, 0, 128, lane);
#pragma unroll
    for (int c = 0; c < 4; ++c) {
      int n = n0 + c * 16 + nn;
      float bv = enc2b[n];
#pragma unroll
      for (int r = 0; r < 8; ++r)
        t2[m0 + hs * 8 + r][n] = (_Float16)fmaxf(acc[c][r] + bv, 0.f);
    }
    __syncthreads();

    // ---- enc_mean / enc_std (+ reparameterized z -> t1) ------------------
    {
      v8f am[4], as[4];
      zero4(am);
      zero4(as);
      mm8(am, as, &t2[0][0], 128, m0, emp, esp, 8, nt0, 128, lane);
#pragma unroll
      for (int c = 0; c < 4; ++c) {
        int n = n0 + c * 16 + nn;
#pragma unroll
        for (int r = 0; r < 8; ++r) {
          int m = m0 + hs * 8 + r;
          float mu = am[c][r] + emb[n];
          float sd = sp_act(as[c][r] + esb[n]);
          out_im[obase + m * 128 + n] = mu;
          out_istd[obase + m * 128 + n] = sd;
          t1[m][n] = (_Float16)(mu + epsT[m * 128 + n] * sd);
        }
      }
    }
    __syncthreads();

    // ---- phi_z: relu(z @ W + b) -> pz16 + dec_in[:,0:128] ----------------
    zero4(acc);
    mm4(acc, &t1[0][0], 128, m0, pzp, 8, nt0, 0, 128, lane);
#pragma unroll
    for (int c = 0; c < 4; ++c) {
      int n = n0 + c * 16 + nn;
      float bv = pzb[n];
#pragma unroll
      for (int r = 0; r < 8; ++r) {
        int m = m0 + hs * 8 + r;
        float v = fmaxf(acc[c][r] + bv, 0.f);
        pz16[m][n] = (_Float16)v;
        dec_in_all[(size_t)(t * 64 + m) * 256 + n] = v;
      }
    }
    __syncthreads();

    // ---- GRU (no bias, gate order r,z,n; n = tanh(i_n + r*h_n)) ----------
    {
      v8f ar[4], az[4], ain[4], ahn[4];
      zero4(ar); zero4(az); zero4(ain); zero4(ahn);
      mm12(ar, az, ain, phiA, 384, m0, gihp, nt0, 0, 384, lane);
      mm12(ar, az, ain, &pz16[0][0], 128, m0, gihp, nt0, 12, 128, lane);
      mm12(ar, az, ahn, &h16[0][0], 128, m0, ghhp, nt0, 0, 128, lane);

      float hnew[4][8];
#pragma unroll
      for (int c = 0; c < 4; ++c) {
        int n = n0 + c * 16 + nn;
#pragma unroll
        for (int r = 0; r < 8; ++r) {
          int m = m0 + hs * 8 + r;
          float rg = sigm(ar[c][r]);
          float zg = sigm(az[c][r]);
          float ng = tanhf(ain[c][r] + rg * ahn[c][r]);
          float hp = (float)h16[m][n];
          h_prev_all[obase + m * 128 + n] = hp;
          dec_in_all[(size_t)(t * 64 + m) * 256 + 128 + n] = hp;
          hnew[c][r] = (1.f - zg) * ng + zg * hp;
        }
      }
      __syncthreads();
#pragma unroll
      for (int c = 0; c < 4; ++c) {
        int n = n0 + c * 16 + nn;
#pragma unroll
        for (int r = 0; r < 8; ++r) h16[m0 + hs * 8 + r][n] = (_Float16)hnew[c][r];
      }
      __syncthreads();
    }
  }
}

// ---------------------------------------------------------------------------
// Host side
// ---------------------------------------------------------------------------
static inline void launch_gemm(hipStream_t s, const float* A, const float* B,
                               const float* bias, float* C, _Float16* C16,
                               int M, int N, int K, int lda, int ldb, int ldc,
                               int act) {
  dim3 grid((N + 127) / 128, (M + 127) / 128), blk(256);
  gemm_bias_act_kernel<<<grid, blk, 0, s>>>(A, B, bias, C, C16, M, N, K, lda,
                                            ldb, ldc, act);
}
static inline void launch_dual(hipStream_t s, const float* A, const float* B0,
                               const float* B1, const float* b0,
                               const float* b1, float* C0, float* C1, int M,
                               int N, int K, int lda, int ldb, int ldc,
                               int act0, int act1) {
  dim3 grid((N + 127) / 128, (M + 127) / 128), blk(256);
  gemm_dual_kernel<<<grid, blk, 0, s>>>(A, B0, B1, b0, b1, C0, C1, M, N, K,
                                        lda, ldb, ldc, act0, act1);
}
static inline void launch_pack(hipStream_t s, const float* B, _Float16* Bp,
                               int K, int N) {
  int total = K * N;
  pack_b_kernel<<<(total + 255) / 256, 256, 0, s>>>(B, Bp, K, N);
}

extern "C" void kernel_launch(void* const* d_in, const int* in_sizes, int n_in,
                              void* d_out, int out_size, void* d_ws,
                              size_t ws_size, hipStream_t stream) {
  (void)in_sizes; (void)n_in; (void)out_size; (void)ws_size;
  // d_in: 0 audio 1 text 2 visual 3 eps, then params in dict insertion order
  // (tuples flattened W,b): phi_audio1..2, phi_text1..2, phi_vis1..2, phi_z,
  // enc1, enc2, enc_mean, enc_std, prior, prior_mean, prior_std,
  // dec_a1..2, dec_a_mean, dec_a_std, dec_t1..2, dec_t_mean, dec_t_std,
  // dec_v1..2, dec_v_mean, dec_v_std, dec_val1..2, dec_val_mean, dec_val_std,
  // gru_ih(64), gru_hh(65)
  const float* audio  = (const float*)d_in[0];
  const float* text   = (const float*)d_in[1];
  const float* visual = (const float*)d_in[2];
  const float* eps    = (const float*)d_in[3];
#define PW(i) ((const float*)d_in[(i)])

  const int M = 64 * 64;  // 4096

  // ---- workspace layout: f32 region then f16 region (~18 MB total) --------
  float* ws      = (float*)d_ws;
  float* tmp     = ws;                        // [4096,128]
  float* h_prev  = tmp + (size_t)M * 128;     // [4096,128]
  float* dec_in  = h_prev + (size_t)M * 128;  // [4096,256]
  float* prior_t = dec_in + (size_t)M * 256;  // [4096,128]
  float* dh1     = prior_t + (size_t)M * 128; // [4096,128]
  float* dh2     = dh1 + (size_t)M * 128;     // [4096,128]
  _Float16* hbase = (_Float16*)(dh2 + (size_t)M * 128);
  _Float16* phi16 = hbase;                    // [4096,384] f16
  _Float16* p_enc1 = phi16 + (size_t)M * 384; // 512x128
  _Float16* p_enc2 = p_enc1 + 512 * 128;      // 128x128
  _Float16* p_em   = p_enc2 + 128 * 128;
  _Float16* p_es   = p_em + 128 * 128;
  _Float16* p_pz   = p_es + 128 * 128;
  _Float16* p_gih  = p_pz + 128 * 128;        // 512x384
  _Float16* p_ghh  = p_gih + 512 * 384;       // 128x384

  // ---- output offsets (reference tuple order) ------------------------------
  float* out     = (float*)d_out;
  float* o_im    = out;
  float* o_istd  = out + (size_t)M * 128;
  float* o_pm    = o_istd + (size_t)M * 128;
  float* o_pstd  = o_pm + (size_t)M * 128;
  float* o_am    = o_pstd + (size_t)M * 128;
  float* o_ast   = o_am + (size_t)M * 990;
  float* o_tm    = o_ast + (size_t)M * 990;
  float* o_tst   = o_tm + (size_t)M * 300;
  float* o_vm    = o_tst + (size_t)M * 300;
  float* o_vst   = o_vm + (size_t)M * 4096;
  float* o_valm  = o_vst + (size_t)M * 4096;
  float* o_valst = o_valm + (size_t)M * 1;

  // ---- pre-swizzle recurrent weights into WMMA fragment order --------------
  launch_pack(stream, PW(18), p_enc1, 512, 128);
  launch_pack(stream, PW(20), p_enc2, 128, 128);
  launch_pack(stream, PW(22), p_em, 128, 128);
  launch_pack(stream, PW(24), p_es, 128, 128);
  launch_pack(stream, PW(16), p_pz, 128, 128);
  launch_pack(stream, PW(64), p_gih, 512, 384);
  launch_pack(stream, PW(65), p_ghh, 128, 384);

  // ---- feature extractors -> phi16 [4096,384] (f16, concat via col offset) -
  launch_gemm(stream, audio, PW(4), PW(5), tmp, nullptr, M, 128, 990, 990, 128, 128, ACT_RELU);
  launch_gemm(stream, tmp, PW(6), PW(7), nullptr, phi16 + 0, M, 128, 128, 128, 128, 384, ACT_RELU);
  launch_gemm(stream, text, PW(8), PW(9), tmp, nullptr, M, 128, 300, 300, 128, 128, ACT_RELU);
  launch_gemm(stream, tmp, PW(10), PW(11), nullptr, phi16 + 128, M, 128, 128, 128, 128, 384, ACT_RELU);
  launch_gemm(stream, visual, PW(12), PW(13), tmp, nullptr, M, 128, 4096, 4096, 128, 128, ACT_RELU);
  launch_gemm(stream, tmp, PW(14), PW(15), nullptr, phi16 + 256, M, 128, 128, 128, 128, 384, ACT_RELU);

  // ---- sequential recurrence (one WGP, 8 waves, WMMA) ----------------------
  vrnn_scan_kernel<<<1, 256, 0, stream>>>(
      phi16, eps, p_enc1, PW(19), p_enc2, PW(21), p_em, PW(23), p_es, PW(25),
      p_pz, PW(17), p_gih, p_ghh, o_im, o_istd, h_prev, dec_in);

  // ---- prior (mean/std fused) ----------------------------------------------
  launch_gemm(stream, h_prev, PW(26), PW(27), prior_t, nullptr, M, 128, 128, 128, 128, 128, ACT_RELU);
  launch_dual(stream, prior_t, PW(28), PW(30), PW(29), PW(31), o_pm, o_pstd,
              M, 128, 128, 128, 128, 128, ACT_NONE, ACT_SP);

  // ---- decoders (dec_in = [phi_z | h_prev]; heads fused mean/std) ----------
  launch_gemm(stream, dec_in, PW(32), PW(33), dh1, nullptr, M, 128, 256, 256, 128, 128, ACT_RELU);
  launch_gemm(stream, dh1, PW(34), PW(35), dh2, nullptr, M, 128, 128, 128, 128, 128, ACT_RELU);
  launch_dual(stream, dh2, PW(36), PW(38), PW(37), PW(39), o_am, o_ast,
              M, 990, 128, 128, 990, 990, ACT_NONE, ACT_SP);

  launch_gemm(stream, dec_in, PW(40), PW(41), dh1, nullptr, M, 128, 256, 256, 128, 128, ACT_RELU);
  launch_gemm(stream, dh1, PW(42), PW(43), dh2, nullptr, M, 128, 128, 128, 128, 128, ACT_RELU);
  launch_dual(stream, dh2, PW(44), PW(46), PW(45), PW(47), o_tm, o_tst,
              M, 300, 128, 128, 300, 300, ACT_NONE, ACT_SP);

  launch_gemm(stream, dec_in, PW(48), PW(49), dh1, nullptr, M, 128, 256, 256, 128, 128, ACT_RELU);
  launch_gemm(stream, dh1, PW(50), PW(51), dh2, nullptr, M, 128, 128, 128, 128, 128, ACT_RELU);
  launch_dual(stream, dh2, PW(52), PW(54), PW(53), PW(55), o_vm, o_vst,
              M, 4096, 128, 128, 4096, 4096, ACT_NONE, ACT_SP);

  launch_gemm(stream, dec_in, PW(56), PW(57), dh1, nullptr, M, 128, 256, 256, 128, 128, ACT_RELU);
  launch_gemm(stream, dh1, PW(58), PW(59), dh2, nullptr, M, 128, 128, 128, 128, 128, ACT_RELU);
  launch_dual(stream, dh2, PW(60), PW(62), PW(61), PW(63), o_valm, o_valst,
              M, 1, 128, 128, 1, 1, ACT_NONE, ACT_SP);
#undef PW
}
